// GRU_61031485276746
// MI455X (gfx1250) — matrix operand order repaired and checked by reading, compiled, not run on hardware
//
#include <hip/hip_runtime.h>
#include <hip/hip_bf16.h>

#define SEQ   2048
#define D     512
#define VOCAB 32000

typedef __attribute__((ext_vector_type(16))) __bf16 v16bf;
typedef __attribute__((ext_vector_type(8)))  float  v8f;

// pointer types for the async global->LDS builtin (per hipcc diagnostic:
// param 0 is 'int __vector(4) __device__ *', i.e. v4i in addrspace(1))
typedef int v4i __attribute__((vector_size(16)));
typedef __attribute__((address_space(1))) v4i* gv4i_p;
typedef __attribute__((address_space(3))) v4i* lv4i_p;

// ----- CDNA5 async global->LDS path (guarded; sync fallback keeps compile safe)
#if defined(__has_builtin)
#if __has_builtin(__builtin_amdgcn_global_load_async_to_lds_b128)
#define HAS_ASYNC_LDS 1
#endif
#endif

#ifdef HAS_ASYNC_LDS
#if __has_builtin(__builtin_amdgcn_s_wait_asynccnt)
#define WAIT_ASYNC(n) __builtin_amdgcn_s_wait_asynccnt(n)
#else
#define WAIT_ASYNC(n) asm volatile("s_wait_asynccnt %0" ::"i"(n) : "memory")
#endif
#else
#define WAIT_ASYNC(n)
#endif

// ---------------------------------------------------------------------------
// Kernel 1: convert wy (fp32, VOCAB x D) -> bf16 in workspace
// ---------------------------------------------------------------------------
__global__ __launch_bounds__(256) void wy_to_bf16(const float* __restrict__ wy,
                                                  __hip_bfloat16* __restrict__ wy_bf) {
    size_t i = (size_t)blockIdx.x * 2048 + threadIdx.x * 8;
    #pragma unroll
    for (int k = 0; k < 8; ++k) {
        wy_bf[i + k] = __float2bfloat16(wy[i + k]);
    }
}

// ---------------------------------------------------------------------------
// Kernel 2: sequential GRU scan. One workgroup (1024 threads, 32 waves) on
// one WGP. Hidden state in LDS; U matrices stream from L2 each step.
// Prefetches the (data-dependent) embedding rows for t+1 one step ahead.
// ---------------------------------------------------------------------------
__global__ __launch_bounds__(1024) void gru_scan(
    const int* __restrict__ X,
    const float* __restrict__ wz, const float* __restrict__ uz,
    const float* __restrict__ w,  const float* __restrict__ u,
    const float* __restrict__ wr, const float* __restrict__ ur,
    const float* __restrict__ bz, const float* __restrict__ br,
    const float* __restrict__ bh,
    float* __restrict__ hs_out, __hip_bfloat16* __restrict__ hs_bf)
{
    __shared__ float h[D];
    __shared__ float zs[D];
    __shared__ float rs[D];
    __shared__ float rh[D];
    __shared__ float part[1024];

    const int tid  = threadIdx.x;
    const int j    = tid & (D - 1);
    const int half = tid >> 9;          // 0: z-gate, 1: r-gate / upper K half

    if (tid < D) h[tid] = 0.0f;
    __syncthreads();

    const float* Ugate = half ? ur : uz;
    const float* Wgate = half ? wr : wz;
    const float* bgate = half ? br : bz;
    const float4* urow  = (const float4*)(Ugate + (size_t)j * D);
    const float4* u2row = (const float4*)(u + (size_t)j * D) + half * (D / 8);

    for (int t = 0; t < SEQ; ++t) {
        const int idx = X[t];

        // prefetch next step's data-dependent embedding rows
        {
            const int idx_n = X[(t + 1 < SEQ) ? (t + 1) : t];
            __builtin_prefetch(Wgate + (size_t)idx_n * D + j, 0, 0);
            if (tid < D) __builtin_prefetch(w + (size_t)idx_n * D + tid, 0, 0);
        }

        // -------- phase 1: z and r matvecs in parallel -------------------
        float e = Wgate[(size_t)idx * D + j];
        float acc = 0.0f;
        const float4* hv = (const float4*)h;
        #pragma unroll 4
        for (int k4 = 0; k4 < D / 4; ++k4) {
            float4 uv = urow[k4];
            float4 hh = hv[k4];
            acc = fmaf(uv.x, hh.x, acc);
            acc = fmaf(uv.y, hh.y, acc);
            acc = fmaf(uv.z, hh.z, acc);
            acc = fmaf(uv.w, hh.w, acc);
        }
        float g = 1.0f / (1.0f + __expf(-(e + acc + bgate[j])));
        if (half) rs[j] = g; else zs[j] = g;
        __syncthreads();

        if (tid < D) rh[tid] = rs[tid] * h[tid];
        __syncthreads();

        // -------- phase 2: candidate matvec, K split across pairs --------
        float acc2 = 0.0f;
        const float4* rhv = (const float4*)rh + half * (D / 8);
        #pragma unroll 4
        for (int k4 = 0; k4 < D / 8; ++k4) {
            float4 uv = u2row[k4];
            float4 rv = rhv[k4];
            acc2 = fmaf(uv.x, rv.x, acc2);
            acc2 = fmaf(uv.y, rv.y, acc2);
            acc2 = fmaf(uv.z, rv.z, acc2);
            acc2 = fmaf(uv.w, rv.w, acc2);
        }
        part[tid] = acc2;
        __syncthreads();

        if (tid < D) {
            float dot = part[tid] + part[tid + D];
            float eh  = w[(size_t)idx * D + tid];
            float hx  = tanhf(eh + dot + bh[tid]);
            float z   = zs[tid];
            float hn  = h[tid] * z + (1.0f - z) * hx;
            h[tid] = hn;
            hs_out[(size_t)t * D + tid] = hn;
            hs_bf[(size_t)t * D + tid]  = __float2bfloat16(hn);
        }
        __syncthreads();
    }
}

// ---------------------------------------------------------------------------
// Kernel 3: logits = hs @ wy^T + by via v_wmma_f32_16x16x32_bf16.
// Block tile 128x128, 256 threads = 8 waves in a 2x4 grid; each wave owns
// 64x32 = 4x2 WMMA accumulators. K=512 in steps of 32, double-buffered LDS
// filled by async global->LDS loads (ASYNCcnt-throttled).
// ---------------------------------------------------------------------------
#define TM 128
#define TN 128
#define TK 32
#define LDK 40   // padded LDS row stride (bf16 elems); 80B keeps 16B alignment

union Frag { v16bf v; float4 f4[2]; };

// stage one 128x32 bf16 tile (2 x 16B chunks per thread)
__device__ __forceinline__ void stage_tile(const __hip_bfloat16* __restrict__ gsrc,
                                           int row_base, int kk,
                                           __bf16 (*lbuf)[LDK], int tid)
{
    #pragma unroll
    for (int i = 0; i < 2; ++i) {
        const int c   = tid * 2 + i;        // 0..511
        const int row = c >> 2;             // 0..127
        const int kc  = (c & 3) * 8;        // 0,8,16,24
        const __hip_bfloat16* g = gsrc + (size_t)(row_base + row) * D + kk + kc;
        __bf16* l = &lbuf[row][kc];
#ifdef HAS_ASYNC_LDS
        __builtin_amdgcn_global_load_async_to_lds_b128(
            (gv4i_p)(void*)g, (lv4i_p)(void*)l, 0, 0);
#else
        *(float4*)l = *(const float4*)g;
#endif
    }
}

__global__ __launch_bounds__(256) void gemm_logits(
    const __hip_bfloat16* __restrict__ hs_bf,
    const __hip_bfloat16* __restrict__ wy_bf,
    const float* __restrict__ by,
    float* __restrict__ logits)
{
    __shared__ __align__(16) __bf16 At[2][TM][LDK];
    __shared__ __align__(16) __bf16 Bt[2][TN][LDK];

    const int tid   = threadIdx.x;
    const int lane  = tid & 31;
    const int wid   = tid >> 5;
    const int wm    = (wid & 1) * 64;     // wave M offset within tile
    const int wn    = (wid >> 1) * 32;    // wave N offset within tile
    const int mbase = blockIdx.y * TM;
    const int nbase = blockIdx.x * TN;

    const int lrow  = lane & 15;          // fragment row/col within 16
    const int lhi   = lane >> 4;          // lane half (0/1)

    v8f acc[4][2];
    #pragma unroll
    for (int i = 0; i < 4; ++i)
        #pragma unroll
        for (int jj = 0; jj < 2; ++jj)
            acc[i][jj] = (v8f){0.f,0.f,0.f,0.f,0.f,0.f,0.f,0.f};

    // prologue: async-stage tile 0 into buffer 0
    stage_tile(hs_bf, mbase, 0, At[0], tid);
    stage_tile(wy_bf, nbase, 0, Bt[0], tid);

    const int NK = D / TK;                // 16
    for (int it = 0; it < NK; ++it) {
        const int cur = it & 1;
        if (it + 1 < NK) {
            // fill the other buffer while we compute on this one
            stage_tile(hs_bf, mbase, (it + 1) * TK, At[cur ^ 1], tid);
            stage_tile(wy_bf, nbase, (it + 1) * TK, Bt[cur ^ 1], tid);
            WAIT_ASYNC(4);                // tile `it`'s 4 loads complete
        } else {
            WAIT_ASYNC(0);
        }
        __syncthreads();

        // A fragments: ISA 16-bit A 16x32 layout
        //   lane<16: row=lane,    K={0..7},{16..23}; lane>=16: K={8..15},{24..31}
        Frag af[4];
        #pragma unroll
        for (int mt = 0; mt < 4; ++mt) {
            const int row = wm + mt * 16 + lrow;
            const int kb  = lhi * 8;
            af[mt].f4[0] = *(const float4*)&At[cur][row][kb];
            af[mt].f4[1] = *(const float4*)&At[cur][row][16 + kb];
        }
        // B fragments: ISA B 32x16 layout (same K order as A)
        //   lane<16: col=lane, K={0..7},{16..23}; lane>=16: col=lane-16, K={8..15},{24..31}
        Frag bfr[2];
        #pragma unroll
        for (int nt = 0; nt < 2; ++nt) {
            const int col = wn + nt * 16 + lrow;
            const int kb  = lhi * 8;
            bfr[nt].f4[0] = *(const float4*)&Bt[cur][col][kb];
            bfr[nt].f4[1] = *(const float4*)&Bt[cur][col][kb + 16];
        }

        #pragma unroll
        for (int mt = 0; mt < 4; ++mt)
            #pragma unroll
            for (int nt = 0; nt < 2; ++nt)
                acc[mt][nt] = __builtin_amdgcn_wmma_f32_16x16x32_bf16(
                    false, af[mt].v, false, bfr[nt].v,
                    (short)0, acc[mt][nt], false, false);

        __syncthreads();
    }

    // epilogue: C/D layout — lane L, VGPR v -> row = v + 8*(L>>4), col = L&15
    #pragma unroll
    for (int nt = 0; nt < 2; ++nt) {
        const int col_g  = nbase + wn + nt * 16 + lrow;
        const float bias = by[col_g];
        #pragma unroll
        for (int mt = 0; mt < 4; ++mt) {
            #pragma unroll
            for (int v = 0; v < 8; ++v) {
                const int row_g = mbase + wm + mt * 16 + v + 8 * lhi;
                logits[(size_t)row_g * VOCAB + col_g] = acc[mt][nt][v] + bias;
            }
        }
    }
}

// ---------------------------------------------------------------------------
// Kernel 4: in-place log_softmax per row of (SEQ, VOCAB).
// ---------------------------------------------------------------------------
__global__ __launch_bounds__(256) void logsoftmax_rows(float* __restrict__ logits) {
    __shared__ float red[256];
    const int tid = threadIdx.x;
    float* row = logits + (size_t)blockIdx.x * VOCAB;

    float m = -3.4e38f;
    for (int i = tid; i < VOCAB; i += 256) m = fmaxf(m, row[i]);
    red[tid] = m;
    __syncthreads();
    for (int s = 128; s > 0; s >>= 1) {
        if (tid < s) red[tid] = fmaxf(red[tid], red[tid + s]);
        __syncthreads();
    }
    const float rmax = red[0];
    __syncthreads();

    float sum = 0.0f;
    for (int i = tid; i < VOCAB; i += 256) sum += __expf(row[i] - rmax);
    red[tid] = sum;
    __syncthreads();
    for (int s = 128; s > 0; s >>= 1) {
        if (tid < s) red[tid] += red[tid + s];
        __syncthreads();
    }
    const float shift = rmax + logf(red[0]);
    __syncthreads();

    for (int i = tid; i < VOCAB; i += 256) row[i] = row[i] - shift;
}

// ---------------------------------------------------------------------------
extern "C" void kernel_launch(void* const* d_in, const int* in_sizes, int n_in,
                              void* d_out, int out_size, void* d_ws, size_t ws_size,
                              hipStream_t stream) {
    const int*   X  = (const int*)  d_in[0];
    const float* wz = (const float*)d_in[1];
    const float* uz = (const float*)d_in[2];
    const float* w  = (const float*)d_in[3];
    const float* u  = (const float*)d_in[4];
    const float* wr = (const float*)d_in[5];
    const float* ur = (const float*)d_in[6];
    const float* wy = (const float*)d_in[7];
    const float* bz = (const float*)d_in[8];
    const float* br = (const float*)d_in[9];
    const float* bh = (const float*)d_in[10];
    const float* by = (const float*)d_in[11];

    float* logits  = (float*)d_out;                       // SEQ*VOCAB
    float* hiddens = logits + (size_t)SEQ * VOCAB;        // SEQ*D

    __hip_bfloat16* wy_bf = (__hip_bfloat16*)d_ws;        // VOCAB*D
    __hip_bfloat16* hs_bf = wy_bf + (size_t)VOCAB * D;    // SEQ*D

    wy_to_bf16<<<dim3((VOCAB * D) / 2048), 256, 0, stream>>>(wy, wy_bf);

    gru_scan<<<1, 1024, 0, stream>>>(X, wz, uz, w, u, wr, ur, bz, br, bh,
                                     hiddens, hs_bf);

    gemm_logits<<<dim3(VOCAB / TN, SEQ / TM), 256, 0, stream>>>(hs_bf, wy_bf,
                                                                by, logits);

    logsoftmax_rows<<<SEQ, 256, 0, stream>>>(logits);
}
